// LeNet5_Orig_C3_61409442398651
// MI455X (gfx1250) — matrix operand order repaired and checked by reading, not compiled, not run on hardware
//
#include <hip/hip_runtime.h>

typedef _Float16 v16h __attribute__((ext_vector_type(16)));
typedef float    v8f  __attribute__((ext_vector_type(8)));
typedef _Float16 half4 __attribute__((ext_vector_type(4)));

// LeNet C3 connectivity, one 6-bit mask per output map
__device__ __constant__ unsigned char CONN_BITS[16] = {
    7, 14, 28, 56, 49, 35,          // 3-input maps
    15, 30, 60, 57, 51, 39, 27, 54, 45,  // 4-input maps
    63                              // 6-input map
};

#define IMG_H  64
#define IMG_W  64
#define LDSW   72      // padded LDS row width (64 data + 8 zero pad)
#define OUT_HW 60
#define NCH    6
#define NOC    16

__global__ __launch_bounds__(256)
void c3_wmma_kernel(const float* __restrict__ x,
                    const float* __restrict__ W,
                    const float* __restrict__ bias,
                    float* __restrict__ out) {
    // f16 image, 6 channels, rows padded to 72 halves => 55,296 bytes
    __shared__ _Float16 smem[NCH * IMG_H * LDSW];

    const int b    = blockIdx.x;
    const int tid  = threadIdx.x;
    const int lane = tid & 31;
    const int wave = tid >> 5;
    const int hi   = (lane & 16) ? 1 : 0;   // which half of the wave

    // ---- zero the pad columns (cols 64..71 of every row) ----
    for (int i = tid; i < NCH * IMG_H * 2; i += 256) {
        int row  = i >> 1;
        int part = i & 1;
        half4 z = {};
        *(half4*)&smem[row * LDSW + IMG_W + part * 4] = z;  // 8B aligned
    }

    // ---- stage the whole image to LDS as f16 (coalesced float4 reads) ----
    const float4* xg = (const float4*)(x + (size_t)b * (NCH * IMG_H * IMG_W));
    for (int q = tid; q < NCH * IMG_H * IMG_W / 4; q += 256) {
        float4 v = xg[q];
        int c   = q >> 10;           // 1024 float4 per 64x64 plane
        int rem = q & 1023;
        int y   = rem >> 4;
        int x4  = (rem & 15) << 2;
        half4 hv;
        hv[0] = (_Float16)v.x; hv[1] = (_Float16)v.y;
        hv[2] = (_Float16)v.z; hv[3] = (_Float16)v.w;
        *(half4*)&smem[(c * IMG_H + y) * LDSW + x4] = hv;   // 8B aligned
    }

    // ---- build masked, zero-padded weight (A) fragments: K' = r*8+kw, r=c*5+kh ----
    // A 16-bit layout: lanes<16 hold K {0..7,16..23}, lanes>=16 hold K {8..15,24..31}
    const int oc = lane & 15;
    const unsigned int cb = CONN_BITS[oc];
    v16h afrag[8];
#pragma unroll
    for (int j = 0; j < 8; ++j) {
        v16h a;
#pragma unroll
        for (int h = 0; h < 16; ++h) {
            int koff = ((h < 8) ? h : (h + 8)) + hi * 8;
            int kp   = 32 * j + koff;
            int r    = kp >> 3;
            int kw   = kp & 7;
            float val = 0.f;
            if (r < 30 && kw < 5) {
                int ci = r / 5;
                int kh = r - ci * 5;
                if ((cb >> ci) & 1)
                    val = W[oc * 150 + ci * 25 + kh * 5 + kw];
            }
            a[h] = (_Float16)val;
        }
        afrag[j] = a;
    }

    // per-lane bias for the 8 C rows this lane owns (M = v + 8*hi)
    float biasv[8];
#pragma unroll
    for (int v = 0; v < 8; ++v) biasv[v] = bias[v + hi * 8];

    __syncthreads();

    // ---- main loop: 240 tiles (60 rows x 4 col-tiles), 8 waves round-robin ----
    const int ncol = lane & 15;
    float* outb = out + (size_t)b * (NOC * OUT_HW * OUT_HW);

    for (int t = wave; t < 240; t += 8) {
        int oy = t >> 2;
        int ts = t & 3;
        int xb = (ts == 3) ? 44 : (ts << 4);   // {0,16,32,44}; cols 44..47 redone
        int ox = xb + ncol;

        v8f acc = {};
#pragma unroll
        for (int j = 0; j < 8; ++j) {
            // B layout: lanes<16 hold K {0..7,16..23}, lanes>=16 hold K {8..15,24..31} of the chunk
            int r0 = 4 * j + hi;
            int r1 = r0 + 2;
            if (r0 > 29) r0 = 29;              // K' >= 240 pads: A is zero there
            if (r1 > 29) r1 = 29;
            int c0 = r0 / 5, kh0 = r0 - c0 * 5;
            int c1 = r1 / 5, kh1 = r1 - c1 * 5;
            const _Float16* p0 = &smem[(c0 * IMG_H + oy + kh0) * LDSW + ox];
            const _Float16* p1 = &smem[(c1 * IMG_H + oy + kh1) * LDSW + ox];
            v16h bf;
#pragma unroll
            for (int kk = 0; kk < 8; ++kk) {
                bf[kk]     = p0[kk];
                bf[8 + kk] = p1[kk];
            }
            acc = __builtin_amdgcn_wmma_f32_16x16x32_f16(
                      false, afrag[j], false, bf, (short)0, acc, false, false);
        }

        // ---- epilogue: C VGPR v -> M = v + 8*hi, N = ox; coalesced f32 stores ----
        float* op = outb + (size_t)(hi * 8) * (OUT_HW * OUT_HW) + oy * OUT_HW + ox;
#pragma unroll
        for (int v = 0; v < 8; ++v) {
            *op = acc[v] + biasv[v];
            op += OUT_HW * OUT_HW;
        }
    }
}

extern "C" void kernel_launch(void* const* d_in, const int* in_sizes, int n_in,
                              void* d_out, int out_size, void* d_ws, size_t ws_size,
                              hipStream_t stream) {
    const float* x    = (const float*)d_in[0];  // [B, 6, 64, 64]
    const float* W    = (const float*)d_in[1];  // [16, 6, 5, 5]
    const float* bias = (const float*)d_in[2];  // [16]
    float* out = (float*)d_out;                 // [B, 16, 60, 60]

    int batch = in_sizes[0] / (NCH * IMG_H * IMG_W);
    c3_wmma_kernel<<<batch, 256, 0, stream>>>(x, W, bias, out);
}
